// HeteroGAT_78391743087101
// MI455X (gfx1250) — matrix operand order repaired and checked by reading, not compiled, not run on hardware
//
#include <hip/hip_runtime.h>
#include <hip/hip_bf16.h>

// ---------------------------------------------------------------------------
// HeteroGAT on MI455X (gfx1250, wave32).
//  - GEMMs via v_wmma_f32_16x16x32_f16; BOTH A and B tiles staged by the
//    Tensor Data Mover (tensor_load_to_lds) with HW LDS padding (80B rows),
//    double-buffered so DMA of tile kb+1 overlaps WMMA on tile kb.
//  - X converted f32->f16 once per layer; W transposed+converted to f16
//    [n][k] once per conv, so the GEMM hot loop has no global loads/cvts.
//  - hd never materialized: a_d = x_dst @ (W_dst . att_dst)  (K x 4 GEMV).
//  - Edge softmax: ordered-uint atomicMax (segment max), atomicAdd (denom),
//    then wave-per-edge gather of hs[src] + f32 atomic scatter into out[dst].
// ---------------------------------------------------------------------------

typedef _Float16 v16h __attribute__((ext_vector_type(16)));
typedef _Float16 h8   __attribute__((ext_vector_type(8)));
typedef float    v8f  __attribute__((ext_vector_type(8)));
typedef float    f4   __attribute__((ext_vector_type(4)));
typedef unsigned u32x4 __attribute__((ext_vector_type(4)));
typedef int      i32x4 __attribute__((ext_vector_type(4)));
typedef int      i32x8 __attribute__((ext_vector_type(8)));

#define TILE_ROWS 128
#define LDX 40   // halves per LDS row (32 + 8 pad) -> 80B stride, 16B aligned
#define LDW 40

#if __has_builtin(__builtin_amdgcn_tensor_load_to_lds)
#define USE_TDM 1
#else
#define USE_TDM 0
#endif

#if USE_TDM
// DMA a tiles x 32-half 2D tile (row stride = strideK halves) into LDS with
// pad_interval=3 (16 DW) / pad_amount=3 (4 DW) => 80B LDS row stride (= LDX).
// Rows beyond rowsLeft (tile-local dim1 bound) are zero-filled by the TDM.
__device__ __forceinline__ void tdm_load_tile(
    const _Float16* g, const _Float16* lds,
    unsigned td0, unsigned rowsLeft, unsigned tileRows, unsigned strideK)
{
    const unsigned ldsOff = (unsigned)(uintptr_t)lds;
    const unsigned long long ga = (unsigned long long)(uintptr_t)g;
    // D# group 0: count=1 | lds_addr | global_addr(57b) | type=2
    u32x4 g0 = { 1u, ldsOff,
                 (unsigned)(ga & 0xFFFFFFFFull),
                 (unsigned)((ga >> 32) & 0x1FFFFFFull) | (2u << 30) };
    // D# group 1: data_size=2B | pad_enable | pad_interval=3 | pad_amount=3
    i32x8 g1;
    g1[0] = (int)((1u << 16) | (1u << 20) | (3u << 22) | (3u << 25));
    g1[1] = (int)((td0 & 0xFFFFu) << 16);                   // tensor_dim0 lo16
    g1[2] = (int)((td0 >> 16) | ((rowsLeft & 0xFFFFu) << 16));
    g1[3] = (int)((rowsLeft >> 16) | (32u << 16));          // tile_dim0 = 32
    g1[4] = (int)(tileRows & 0xFFFFu);                      // tile_dim1
    g1[5] = (int)strideK;                                   // dim0_stride lo32
    g1[6] = 0;
    g1[7] = 0;
    i32x4 gz4 = { 0, 0, 0, 0 };
#if defined(__clang_major__) && __clang_major__ >= 23
    i32x8 gz8 = { 0, 0, 0, 0, 0, 0, 0, 0 };
    __builtin_amdgcn_tensor_load_to_lds(g0, g1, gz4, gz4, gz8, 0);
#else
    __builtin_amdgcn_tensor_load_to_lds(g0, g1, gz4, gz4, 0);
#endif
}
#endif

// ----------------------- f32 -> f16 bulk convert ----------------------------
__global__ __launch_bounds__(256) void cvt_f16_k(
    const float* __restrict__ x, _Float16* __restrict__ y, int n8)
{
    const int id = blockIdx.x * 256 + threadIdx.x;
    if (id >= n8) return;
    const float* p = x + (size_t)id * 8;
    f4 a = *(const f4*)p, b = *(const f4*)(p + 4);
    h8 o;
#pragma unroll
    for (int j = 0; j < 4; ++j) { o[j] = (_Float16)a[j]; o[4 + j] = (_Float16)b[j]; }
    *(h8*)(y + (size_t)id * 8) = o;
}

// ---------------- W (Kx256 f32) -> Wt (256xK f16, [n][k]) -------------------
__global__ __launch_bounds__(256) void wtr_k(
    const float* __restrict__ W, _Float16* __restrict__ Wt, int K)
{
    const int id = blockIdx.x * 256 + threadIdx.x;   // id = k*256 + n
    if (id >= K * 256) return;
    const int k = id >> 8, n = id & 255;
    Wt[(size_t)n * K + k] = (_Float16)W[id];
}

// ---------- WMMA GEMM: Y(Nx256,f32) = Xh(NxK,f16) * Wt(256xK,f16)^T ---------
__global__ __launch_bounds__(256) void gemm_wmma_f16(
    const _Float16* __restrict__ Xh, const _Float16* __restrict__ Wt,
    float* __restrict__ Y, int Nrows, int K)
{
    __shared__ __align__(16) _Float16 lsX[2][TILE_ROWS * LDX];
    __shared__ __align__(16) _Float16 lsW[2][64 * LDW];

    const int t    = threadIdx.x;
    const int wave = t >> 5;
    const int lane = t & 31;
    const int rowTile = blockIdx.x * TILE_ROWS;
    const int colBase = blockIdx.y * 64;
    const unsigned rowsLeft = (unsigned)(Nrows - rowTile);

    v8f acc[4];
#pragma unroll
    for (int i = 0; i < 4; ++i)
#pragma unroll
        for (int j = 0; j < 8; ++j) acc[i][j] = 0.0f;

    const int nkb = K >> 5;

#if USE_TDM
    if (wave == 0) {
        tdm_load_tile(Xh + (size_t)rowTile * K, &lsX[0][0],
                      (unsigned)K, rowsLeft, TILE_ROWS, (unsigned)K);
        tdm_load_tile(Wt + (size_t)colBase * K, &lsW[0][0],
                      (unsigned)K, 64u, 64u, (unsigned)K);
        __builtin_amdgcn_s_wait_tensorcnt(0);
    }
    __syncthreads();
#endif

    for (int kb = 0; kb < nkb; ++kb) {
        const int cur = kb & 1;
        const int kG  = kb << 5;

#if USE_TDM
        // prefetch next K-tile into the other buffer while we compute
        if (wave == 0 && kb + 1 < nkb) {
            const int kN = (kb + 1) << 5;
            tdm_load_tile(Xh + (size_t)rowTile * K + kN, &lsX[cur ^ 1][0],
                          (unsigned)(K - kN), rowsLeft, TILE_ROWS, (unsigned)K);
            tdm_load_tile(Wt + (size_t)colBase * K + kN, &lsW[cur ^ 1][0],
                          (unsigned)(K - kN), 64u, 64u, (unsigned)K);
        }
#else
        // fallback: manual f16 staging
        {
            const int r  = t >> 1;
            const int k0 = (t & 1) << 4;
            const int gr = rowTile + r;
            h8 h0, h1;
            if (gr < Nrows) {
                const _Float16* p = Xh + (size_t)gr * K + kG + k0;
                h0 = *(const h8*)p;
                h1 = *(const h8*)(p + 8);
            } else {
#pragma unroll
                for (int j = 0; j < 8; ++j) { h0[j] = (_Float16)0.f; h1[j] = (_Float16)0.f; }
            }
            *(h8*)&lsX[cur][r * LDX + k0]     = h0;
            *(h8*)&lsX[cur][r * LDX + k0 + 8] = h1;
            const int n   = t & 63;
            const int kw0 = (t >> 6) << 3;
            *(h8*)&lsW[cur][n * LDW + kw0] =
                *(const h8*)(Wt + (size_t)(colBase + n) * K + kG + kw0);
        }
        __syncthreads();
#endif

        // ---- WMMA: each wave computes 16 rows x 64 cols ----
        const int m  = lane & 15;
        const int hi = lane >> 4;
        const int rowL = wave * 16 + m;
        v16h A;
        {
            h8 a0 = *(const h8*)&lsX[cur][rowL * LDX + hi * 8];       // K=hi*8..+7
            h8 a1 = *(const h8*)&lsX[cur][rowL * LDX + 16 + hi * 8];  // K=16+hi*8..
#pragma unroll
            for (int j = 0; j < 8; ++j) { A[j] = a0[j]; A[8 + j] = a1[j]; }
        }
#pragma unroll
        for (int nt = 0; nt < 4; ++nt) {
            const int nIdx = nt * 16 + m;
            h8 b0 = *(const h8*)&lsW[cur][nIdx * LDW + hi * 8];
            h8 b1 = *(const h8*)&lsW[cur][nIdx * LDW + 16 + hi * 8];
            v16h B;
#pragma unroll
            for (int j = 0; j < 8; ++j) { B[j] = b0[j]; B[8 + j] = b1[j]; }
            acc[nt] = __builtin_amdgcn_wmma_f32_16x16x32_f16(
                false, A, false, B, (short)0, acc[nt], false, false);
        }

#if USE_TDM
        if (wave == 0 && kb + 1 < nkb) __builtin_amdgcn_s_wait_tensorcnt(0);
#endif
        __syncthreads();
    }

    // ---- store C: lane col = lane&15 ; vgpr v row = v + hi*8 ----
    const int m  = lane & 15;
    const int hi = lane >> 4;
#pragma unroll
    for (int nt = 0; nt < 4; ++nt) {
        const int col = colBase + nt * 16 + m;
#pragma unroll
        for (int v = 0; v < 8; ++v) {
            const int gr = rowTile + wave * 16 + v + hi * 8;
            if (gr < Nrows) Y[(size_t)gr * 256 + col] = acc[nt][v];
        }
    }
}

// ------------------ a_s[n,h] = sum_c hs[n,h*64+c] * att[h,c] ----------------
__global__ __launch_bounds__(256) void row_att_k(
    const float* __restrict__ h, const float* __restrict__ att,
    float* __restrict__ out, int N)
{
    const int wave = threadIdx.x >> 5, lane = threadIdx.x & 31;
    const int row = blockIdx.x * 8 + wave;
    if (row >= N) return;
    const float* p = h + (size_t)row * 256 + lane * 8;
    const float* a = att + lane * 8;
    f4 v0 = *(const f4*)p,       v1 = *(const f4*)(p + 4);
    f4 a0 = *(const f4*)a,       a1 = *(const f4*)(a + 4);
    float s = 0.f;
#pragma unroll
    for (int j = 0; j < 4; ++j) s += v0[j] * a0[j] + v1[j] * a1[j];
    s += __shfl_xor(s, 1);
    s += __shfl_xor(s, 2);
    s += __shfl_xor(s, 4);           // 8 lanes per head
    if ((lane & 7) == 0) out[row * 4 + (lane >> 3)] = s;
}

// -------------- W_eff[k,h] = sum_c W_dst[k,h*64+c] * att_dst[h,c] -----------
__global__ void weff_k(const float* __restrict__ Wd, const float* __restrict__ att,
                       float* __restrict__ weff, int K)
{
    const int id = blockIdx.x * blockDim.x + threadIdx.x;
    if (id >= K * 4) return;
    const int k = id >> 2, hh = id & 3;
    const float* wp = Wd + (size_t)k * 256 + hh * 64;
    const float* ap = att + hh * 64;
    float s = 0.f;
#pragma unroll 8
    for (int c = 0; c < 64; ++c) s += wp[c] * ap[c];
    weff[k * 4 + hh] = s;
}

// ------------------------ a_d[n,h] = x[n,:] @ W_eff[:,h] --------------------
__global__ __launch_bounds__(256) void ad_k(
    const float* __restrict__ X, const float* __restrict__ weff,
    float* __restrict__ ad, int N, int K)
{
    const int wave = threadIdx.x >> 5, lane = threadIdx.x & 31;
    const int row = blockIdx.x * 8 + wave;
    if (row >= N) return;
    const int per = K >> 5;                       // 4 (K=128) or 8 (K=256)
    const float* xp = X + (size_t)row * K + lane * per;
    float acc[4] = {0.f, 0.f, 0.f, 0.f};
    for (int j = 0; j < per; ++j) {
        const float xv = xp[j];
        const float* wp = weff + (size_t)(lane * per + j) * 4;
#pragma unroll
        for (int hh = 0; hh < 4; ++hh) acc[hh] += xv * wp[hh];
    }
#pragma unroll
    for (int hh = 0; hh < 4; ++hh) {
        float s = acc[hh];
        for (int off = 1; off < 32; off <<= 1) s += __shfl_xor(s, off);
        acc[hh] = s;
    }
    if (lane == 0) {
#pragma unroll
        for (int hh = 0; hh < 4; ++hh) ad[row * 4 + hh] = acc[hh];
    }
}

// ------------------- ordered-uint encoding for float atomicMax --------------
__device__ __forceinline__ unsigned enc_f(float f) {
    unsigned u = __float_as_uint(f);
    return (u & 0x80000000u) ? ~u : (u | 0x80000000u);
}
__device__ __forceinline__ float dec_f(unsigned e) {
    unsigned u = (e & 0x80000000u) ? (e & 0x7FFFFFFFu) : ~e;
    return __uint_as_float(u);
}
__device__ __forceinline__ float leaky(float v) { return v > 0.f ? v : 0.2f * v; }

__global__ void init_md_k(unsigned* __restrict__ menc, float* __restrict__ den, int n) {
    const int id = blockIdx.x * blockDim.x + threadIdx.x;
    if (id < n) { menc[id] = 0u; den[id] = 0.f; }
}

__global__ void init_bias_k(float* __restrict__ out, const float* __restrict__ b1,
                            const float* __restrict__ b2, int N)
{
    const size_t id = (size_t)blockIdx.x * 256 + threadIdx.x;
    if (id >= (size_t)N * 256) return;
    const int j = (int)(id & 255);
    out[id] = b1[j] + (b2 ? b2[j] : 0.f);
}

__global__ void edge_max_k(const int* __restrict__ src, const int* __restrict__ dst,
                           const float* __restrict__ as_, const float* __restrict__ ad_,
                           unsigned* __restrict__ menc, int E)
{
    const int id = blockIdx.x * blockDim.x + threadIdx.x;
    if (id >= E * 4) return;
    const int e = id >> 2, hh = id & 3;
    const int s = src[e], d = dst[e];
    const float ev = leaky(as_[s * 4 + hh] + ad_[d * 4 + hh]);
    atomicMax(menc + ((size_t)d * 4 + hh), enc_f(ev));
}

__global__ void edge_den_k(const int* __restrict__ src, const int* __restrict__ dst,
                           const float* __restrict__ as_, const float* __restrict__ ad_,
                           const unsigned* __restrict__ menc, float* __restrict__ den, int E)
{
    const int id = blockIdx.x * blockDim.x + threadIdx.x;
    if (id >= E * 4) return;
    const int e = id >> 2, hh = id & 3;
    const int s = src[e], d = dst[e];
    const float ev = leaky(as_[s * 4 + hh] + ad_[d * 4 + hh]);
    const unsigned me = menc[(size_t)d * 4 + hh];
    const float m = me ? dec_f(me) : 0.f;
    atomicAdd(den + ((size_t)d * 4 + hh), __expf(ev - m));
}

// wave per edge: gather 256-float hs[src], scale by alpha(head), scatter-add
__global__ __launch_bounds__(256) void edge_sc_k(
    const int* __restrict__ src, const int* __restrict__ dst,
    const float* __restrict__ as_, const float* __restrict__ ad_,
    const unsigned* __restrict__ menc, const float* __restrict__ den,
    const float* __restrict__ hs, float* __restrict__ out, int E)
{
    const int wave = threadIdx.x >> 5, lane = threadIdx.x & 31;
    const int e = blockIdx.x * 8 + wave;
    if (e >= E) return;
    const int s = src[e], d = dst[e];
    const int hh = lane >> 3;                       // head for this lane's channels
    const float ev = leaky(as_[s * 4 + hh] + ad_[d * 4 + hh]);
    const unsigned me = menc[(size_t)d * 4 + hh];
    const float m = me ? dec_f(me) : 0.f;
    const float alpha = __expf(ev - m) / (den[(size_t)d * 4 + hh] + 1e-16f);
    const float* hp = hs + (size_t)s * 256 + lane * 8;
    float* op       = out + (size_t)d * 256 + lane * 8;
    f4 v0 = *(const f4*)hp, v1 = *(const f4*)(hp + 4);
#pragma unroll
    for (int j = 0; j < 4; ++j) atomicAdd(op + j, alpha * v0[j]);
#pragma unroll
    for (int j = 0; j < 4; ++j) atomicAdd(op + 4 + j, alpha * v1[j]);
}

__global__ void relu_k(float* __restrict__ x, int n) {
    const int id = blockIdx.x * blockDim.x + threadIdx.x;
    if (id < n) x[id] = fmaxf(x[id], 0.f);
}

// ---------------------------------------------------------------------------
static void run_conv(hipStream_t stream,
    const _Float16* xh_src, int Nsrc, int Ksrc,
    const float* x_dst, int Ndst, int Kdst,
    const int* esrc, const int* edst, int E,
    const float* Wsrc, const float* Wdst, const float* attS, const float* attD,
    _Float16* wt, float* hs, float* as_, float* ad_, unsigned* menc, float* den,
    float* weff, float* out)
{
    wtr_k<<<(Ksrc * 256 + 255) / 256, 256, 0, stream>>>(Wsrc, wt, Ksrc);
    dim3 gGemm((Nsrc + TILE_ROWS - 1) / TILE_ROWS, 4);
    gemm_wmma_f16<<<gGemm, 256, 0, stream>>>(xh_src, wt, hs, Nsrc, Ksrc);
    row_att_k<<<(Nsrc + 7) / 8, 256, 0, stream>>>(hs, attS, as_, Nsrc);
    weff_k<<<(Kdst * 4 + 255) / 256, 256, 0, stream>>>(Wdst, attD, weff, Kdst);
    ad_k<<<(Ndst + 7) / 8, 256, 0, stream>>>(x_dst, weff, ad_, Ndst, Kdst);
    init_md_k<<<(Ndst * 4 + 255) / 256, 256, 0, stream>>>(menc, den, Ndst * 4);
    edge_max_k<<<(E * 4 + 255) / 256, 256, 0, stream>>>(esrc, edst, as_, ad_, menc, E);
    edge_den_k<<<(E * 4 + 255) / 256, 256, 0, stream>>>(esrc, edst, as_, ad_, menc, den, E);
    edge_sc_k<<<(E + 7) / 8, 256, 0, stream>>>(esrc, edst, as_, ad_, menc, den, hs, out, E);
}

extern "C" void kernel_launch(void* const* d_in, const int* in_sizes, int n_in,
                              void* d_out, int out_size, void* d_ws, size_t ws_size,
                              hipStream_t stream)
{
    (void)in_sizes; (void)n_in; (void)out_size; (void)ws_size;

    const int NP = 100000, NA = 50000, E = 500000;
    const float* x_paper  = (const float*)d_in[0];
    const float* x_author = (const float*)d_in[1];
    const int* pp_src = (const int*)d_in[2];
    const int* pp_dst = (const int*)d_in[3];
    const int* ap_src = (const int*)d_in[4];
    const int* ap_dst = (const int*)d_in[5];
    const int* pa_src = (const int*)d_in[6];
    const int* pa_dst = (const int*)d_in[7];

    // workspace layout (float units)
    float* ws = (float*)d_ws;
    float* XPA = ws;                        // 100000*256
    float* XAA = ws + 25600000;             //  50000*256
    float* XPB = ws + 38400000;             // 100000*256
    float* XAB = ws + 64000000;             //  50000*256
    float* HS  = ws + 76800000;             // 100000*256 (hs scratch)
    float* AS  = ws + 102400000;            // 100000*4
    float* AD  = ws + 102800000;            // 100000*4
    unsigned* MENC = (unsigned*)(ws + 103200000); // 100000*4
    float* DEN  = ws + 103600000;           // 100000*4
    float* WEFF = ws + 104000000;           // 256*4
    _Float16* XHP = (_Float16*)(ws + 104004096);  // 100000*256 halves
    _Float16* XHA = (_Float16*)(ws + 116804096);  //  50000*256 halves
    _Float16* WT  = (_Float16*)(ws + 123204096);  //    256*256 halves

    float* outP = (float*)d_out;
    float* outA = outP + (size_t)NP * 256;

    const float* xp = x_paper;  int Kp = 256;
    const float* xa = x_author; int Ka = 128;

    for (int li = 0; li < 3; ++li) {
        // pytree leaf order per layer: ap(0),pa(1),pp(2); items:
        // 0=W_dst 1=W_src 2=att_dst 3=att_src 4=bias
        auto P = [&](int conv, int item) -> const float* {
            return (const float*)d_in[8 + li * 15 + conv * 5 + item];
        };
        float* np = (li == 2) ? outP : ((li & 1) ? XPA : XPB);
        float* na = (li == 2) ? outA : ((li & 1) ? XAA : XAB);

        // convert this layer's node features to f16 once
        {
            const int n8p = (NP * Kp) / 8;
            cvt_f16_k<<<(n8p + 255) / 256, 256, 0, stream>>>(xp, XHP, n8p);
            const int n8a = (NA * Ka) / 8;
            cvt_f16_k<<<(n8a + 255) / 256, 256, 0, stream>>>(xa, XHA, n8a);
        }

        // paper destination: sum of pp-conv and ap-conv (both biases up front)
        init_bias_k<<<NP, 256, 0, stream>>>(np, P(2, 4), P(0, 4), NP);
        run_conv(stream, XHP, NP, Kp, xp, NP, Kp, pp_src, pp_dst, E,
                 P(2, 1), P(2, 0), P(2, 3), P(2, 2),
                 WT, HS, AS, AD, MENC, DEN, WEFF, np);
        run_conv(stream, XHA, NA, Ka, xp, NP, Kp, ap_src, ap_dst, E,
                 P(0, 1), P(0, 0), P(0, 3), P(0, 2),
                 WT, HS, AS, AD, MENC, DEN, WEFF, np);

        // author destination: pa-conv only
        init_bias_k<<<NA, 256, 0, stream>>>(na, P(1, 4), nullptr, NA);
        run_conv(stream, XHP, NP, Kp, xa, NA, Ka, pa_src, pa_dst, E,
                 P(1, 1), P(1, 0), P(1, 3), P(1, 2),
                 WT, HS, AS, AD, MENC, DEN, WEFF, na);

        if (li < 2) {
            relu_k<<<(NP * 256 + 255) / 256, 256, 0, stream>>>(np, NP * 256);
            relu_k<<<(NA * 256 + 255) / 256, 256, 0, stream>>>(na, NA * 256);
        }
        xp = np; Kp = 256;
        xa = na; Ka = 256;
    }
}
